// STSBaselineNet_11544872091747
// MI455X (gfx1250) — matrix operand order repaired and checked by reading, not compiled, not run on hardware
//
#include <hip/hip_runtime.h>
#include <cstdint>
#include <cstddef>

// ---------------- types ----------------
typedef __bf16 bf16;
typedef __attribute__((ext_vector_type(16))) __bf16 v16bf;
typedef __attribute__((ext_vector_type(8)))  __bf16 v8bf;
typedef __attribute__((ext_vector_type(8)))  float  v8f;

// ---------------- problem constants ----------------
#define EDIM   300
#define EPAD   320        // K padded to multiple of 32
#define HDIM   256
#define G4H    1024       // 4*H
#define NB     256        // batch
#define NT     64         // time steps
#define NTOK   (NB*NT)    // 16384
#define NDIR2  2048       // 2 * 4H (both directions concatenated on N)
#define NEGBIG (-1e9f)

// ---------------- gfx1250 async global->LDS copy helpers ----------------
// Generic pointers to __shared__ carry the LDS byte offset in their low 32 bits
// (the flat LDS aperture uses the low 32 address bits).
__device__ inline uint32_t lds_off(const void* p) {
  return (uint32_t)(uintptr_t)p;
}

// 16B per-lane async copy: LDS[lds] = MEM[gaddr]  (tracked by ASYNCcnt)
__device__ inline void async_copy16(uint32_t lds, const void* gaddr) {
  asm volatile("global_load_async_to_lds_b128 %0, %1, off"
               :: "v"(lds), "v"(gaddr)
               : "memory");
}
__device__ inline void wait_async0() {
  asm volatile("s_wait_asynccnt 0" ::: "memory");
}

// ---------------- WMMA fragment helpers (wave32, 16x16x32 bf16) ----------------
__device__ inline v16bf combine8(v8bf lo, v8bf hi) {
  v16bf r;
#pragma unroll
  for (int i = 0; i < 8; ++i) { r[i] = lo[i]; r[i + 8] = hi[i]; }
  return r;
}

// A (16x32, MxK): lane holds row M=lane%16; K chunks at (lane/16)*8 and +16.
__device__ inline v16bf load_a_frag(const bf16* rowbase, int lane, int k) {
  const int sel = (lane >> 4) * 8;
  v8bf lo = *(const v8bf*)(rowbase + k + sel);
  v8bf hi = *(const v8bf*)(rowbase + k + sel + 16);
  return combine8(lo, hi);
}

// B (32x16, KxN) fed as W rows (row N, contiguous K): lane holds W-row N=lane%16,
// K chunks at (lane/16)*8 and +16.  => computes x @ W^T.
__device__ inline v16bf load_b_frag(const bf16* __restrict__ rowbase, int lane, int k) {
  const int sel = (lane >> 4) * 8;
  v8bf lo = *(const v8bf*)(rowbase + k + sel);
  v8bf hi = *(const v8bf*)(rowbase + k + sel + 16);
  return combine8(lo, hi);
}

__device__ inline v8f wmma_bf16(v16bf a, v16bf b, v8f c) {
  return __builtin_amdgcn_wmma_f32_16x16x32_bf16(false, a, false, b, (short)0, c,
                                                 false, false);
}

__device__ inline float sigf(float x) { return 1.0f / (1.0f + __expf(-x)); }

// ---------------- kernel 1: state init ----------------
__global__ void k_init(bf16* __restrict__ hbuf, float* __restrict__ cbuf,
                       float* __restrict__ out) {
  int i = blockIdx.x * blockDim.x + threadIdx.x;
  if (i < 2 * NB * HDIM) {      // == NB * 2H, same count for all three
    hbuf[i] = (bf16)0.0f;
    cbuf[i] = 0.0f;
    out[i]  = NEGBIG;
  }
}

// ---------------- kernel 2: weight/bias conversion to bf16, padded ----------------
__global__ void k_convert(const float* __restrict__ wih_f, const float* __restrict__ whh_f,
                          const float* __restrict__ bf_,   const float* __restrict__ wih_b,
                          const float* __restrict__ whh_b, const float* __restrict__ bb_,
                          bf16* __restrict__ Wih, bf16* __restrict__ Whh,
                          float* __restrict__ bias) {
  const int NW1 = NDIR2 * EPAD;   // Wih: [2048][320]
  const int NW2 = NDIR2 * HDIM;   // Whh: [2048][256]
  int i = blockIdx.x * blockDim.x + threadIdx.x;
  if (i < NW1) {
    int n = i / EPAD, k = i - n * EPAD;
    const float* src = (n < G4H) ? wih_f : wih_b;
    int nn = n & (G4H - 1);
    Wih[i] = (k < EDIM) ? (bf16)src[nn * EDIM + k] : (bf16)0.0f;
  } else if (i < NW1 + NW2) {
    int j = i - NW1;
    int n = j / HDIM, k = j - n * HDIM;
    const float* src = (n < G4H) ? whh_f : whh_b;
    Whh[j] = (bf16)src[(n & (G4H - 1)) * HDIM + k];
  } else if (i < NW1 + NW2 + NDIR2) {
    int n = i - NW1 - NW2;
    bias[n] = (n < G4H) ? bf_[n] : bb_[n - G4H];
  }
}

// ---------------- kernel 3: embedding gather -> bf16, time-major rows ----------------
// X row index = t*NB + b  (token_ids is [B][T] row-major)
__global__ void k_embed(const int* __restrict__ tok, const float* __restrict__ emb,
                        bf16* __restrict__ X) {
  int flat = blockIdx.x;              // b*NT + t
  int b = flat / NT, t = flat - b * NT;
  int id = tok[flat];
  const float* src = emb + (size_t)id * EDIM;
  bf16* dst = X + (size_t)(t * NB + b) * EPAD;
  for (int i = threadIdx.x; i < EPAD; i += blockDim.x)
    dst[i] = (i < EDIM) ? (bf16)src[i] : (bf16)0.0f;
}

// ---------------- kernel 4: input projection GEMM ----------------
// Z[m][n] = sum_k X[m][k] * Wih[n][k] + bias[n],  M=16384, N=2048, K=320
// Block: one 16-row A tile (async-staged to LDS, shared by 8 waves) x 512 N.
__global__ __launch_bounds__(256)
void k_inproj(const bf16* __restrict__ X, const bf16* __restrict__ Wih,
              const float* __restrict__ bias, float* __restrict__ Z) {
  __shared__ __align__(16) bf16 sA[16 * EPAD];    // 10 KB, contiguous 16 rows of X

  const int lane = threadIdx.x & 31;
  const int wave = threadIdx.x >> 5;
  int mTile   = blockIdx.x >> 2;    // 0..1023
  int quarter = blockIdx.x & 3;     // 0..3
  int m0 = mTile * 16;
  int n0 = (quarter * 8 + wave) * 64;

  // stage A tile: 16*320*2 = 10240 B = 640 chunks of 16 B (rows are contiguous)
  {
    const bf16* gsrc = X + (size_t)m0 * EPAD;
    uint32_t base = lds_off(sA);
    for (int c = threadIdx.x; c < 640; c += 256)
      async_copy16(base + c * 16, gsrc + c * 8);
    wait_async0();
  }
  __syncthreads();

  const bf16* arow = sA + (lane & 15) * EPAD;     // LDS reads -> ds_load_b128
  const bf16* brow[4];
#pragma unroll
  for (int nt = 0; nt < 4; ++nt)
    brow[nt] = Wih + (size_t)(n0 + nt * 16 + (lane & 15)) * EPAD;

  v8f acc[4] = {};
  for (int k = 0; k < EPAD; k += 32) {
    __builtin_prefetch(brow[0] + k + 64, 0, 1);   // global_prefetch_b8
    v16bf a = load_a_frag(arow, lane, k);
#pragma unroll
    for (int nt = 0; nt < 4; ++nt) {
      v16bf b = load_b_frag(brow[nt], lane, k);
      acc[nt] = wmma_bf16(a, b, acc[nt]);
    }
  }

  int rbase = m0 + ((lane >> 4) << 3);            // C/D: M = r + 8*(lane/16)
#pragma unroll
  for (int nt = 0; nt < 4; ++nt) {
    int n = n0 + nt * 16 + (lane & 15);
    float bv = bias[n];
#pragma unroll
    for (int r = 0; r < 8; ++r)
      Z[(size_t)(rbase + r) * NDIR2 + n] = acc[nt][r] + bv;
  }
}

// ---------------- kernel 5: per-step recurrent GEMM ----------------
// zr[dir][b][n] = Z[ta*NB+b][dir*1024+n] + sum_k h[dir][b][k]*Whh[dir*1024+n][k]
// Block: one 16-row h tile (async-staged to LDS) x 512 N, per direction.
__global__ __launch_bounds__(256)
void k_recur_gemm(const bf16* __restrict__ hbuf, const bf16* __restrict__ Whh,
                  const float* __restrict__ Z, float* __restrict__ zr, int t) {
  __shared__ __align__(16) bf16 sH[16 * HDIM];    // 8 KB, contiguous 16 rows of h

  const int lane = threadIdx.x & 31;
  const int wave = threadIdx.x >> 5;
  int blk = blockIdx.x;             // 0..63
  int dir = blk >> 5;               // 0 fwd, 1 bwd
  int r5  = blk & 31;
  int mTile = r5 >> 1;              // 0..15
  int half  = r5 & 1;               // 0..1
  int m0 = mTile * 16;
  int n0 = (half * 8 + wave) * 64;

  // stage h tile: 16*256*2 = 8192 B = 512 chunks of 16 B
  {
    const bf16* gsrc = hbuf + (size_t)dir * (NB * HDIM) + (size_t)m0 * HDIM;
    uint32_t base = lds_off(sH);
    for (int c = threadIdx.x; c < 512; c += 256)
      async_copy16(base + c * 16, gsrc + c * 8);
    wait_async0();
  }
  __syncthreads();

  const bf16* arow  = sH + (lane & 15) * HDIM;
  const bf16* wbase = Whh + (size_t)dir * (G4H * HDIM);
  const bf16* brow[4];
#pragma unroll
  for (int nt = 0; nt < 4; ++nt)
    brow[nt] = wbase + (size_t)(n0 + nt * 16 + (lane & 15)) * HDIM;

  v8f acc[4] = {};
  for (int k = 0; k < HDIM; k += 32) {
    v16bf a = load_a_frag(arow, lane, k);
#pragma unroll
    for (int nt = 0; nt < 4; ++nt) {
      v16bf b = load_b_frag(brow[nt], lane, k);
      acc[nt] = wmma_bf16(a, b, acc[nt]);
    }
  }

  int ta = dir ? (NT - 1 - t) : t;  // backward scans reversed time
  const float* zin = Z + (size_t)(ta * NB) * NDIR2 + (size_t)dir * G4H;
  float* zout = zr + (size_t)dir * (NB * G4H);
  int rbase = m0 + ((lane >> 4) << 3);
#pragma unroll
  for (int nt = 0; nt < 4; ++nt) {
    int n = n0 + nt * 16 + (lane & 15);
#pragma unroll
    for (int r = 0; r < 8; ++r) {
      int m = rbase + r;
      zout[(size_t)m * G4H + n] = acc[nt][r] + zin[(size_t)m * NDIR2 + n];
    }
  }
}

// ---------------- kernel 6: gates + state update + fused running max ----------------
__global__ void k_gates(const float* __restrict__ zr, const int* __restrict__ lengths,
                        float* __restrict__ cbuf, bf16* __restrict__ hbuf,
                        float* __restrict__ out, int t) {
  int i = blockIdx.x * blockDim.x + threadIdx.x;   // [0, 2*NB*HDIM)
  if (i >= 2 * NB * HDIM) return;
  int dir = i >> 16;          // 65536 = NB*HDIM per direction
  int rem = i & 65535;
  int b = rem >> 8;
  int n = rem & 255;

  const float* z = zr + (size_t)dir * (NB * G4H) + (size_t)b * G4H;
  float ig = sigf(z[n]);
  float fg = sigf(z[n + HDIM]);
  float gg = tanhf(z[n + 2 * HDIM]);
  float og = sigf(z[n + 3 * HDIM]);

  float c0 = cbuf[i];
  float cn = fg * c0 + ig * gg;
  float hn = og * tanhf(cn);

  int ta = dir ? (NT - 1 - t) : t;
  if (ta < lengths[b]) {      // masked steps leave (h,c) unchanged
    cbuf[i] = cn;
    hbuf[i] = (bf16)hn;
    float* op = out + (size_t)b * (2 * HDIM) + dir * HDIM + n;
    *op = fmaxf(*op, hn);
  }
}

// ---------------- launcher ----------------
extern "C" void kernel_launch(void* const* d_in, const int* in_sizes, int n_in,
                              void* d_out, int out_size, void* d_ws, size_t ws_size,
                              hipStream_t stream) {
  const int*   tok   = (const int*)d_in[0];
  const int*   lens  = (const int*)d_in[1];
  const float* emb   = (const float*)d_in[2];
  const float* wih_f = (const float*)d_in[3];
  const float* whh_f = (const float*)d_in[4];
  const float* b_f   = (const float*)d_in[5];
  const float* wih_b = (const float*)d_in[6];
  const float* whh_b = (const float*)d_in[7];
  const float* b_b   = (const float*)d_in[8];
  float* out = (float*)d_out;

  char* ws = (char*)d_ws;
  size_t off = 0;
  auto walloc = [&](size_t bytes) -> void* {
    void* p = ws + off;
    off = (off + bytes + 255) & ~(size_t)255;
    return p;
  };
  bf16*  X    = (bf16*) walloc((size_t)NTOK * EPAD * sizeof(bf16));   // ~10.5 MB
  bf16*  Wih  = (bf16*) walloc((size_t)NDIR2 * EPAD * sizeof(bf16));  // ~1.3 MB
  bf16*  Whh  = (bf16*) walloc((size_t)NDIR2 * HDIM * sizeof(bf16));  // ~1.0 MB
  float* bias = (float*)walloc((size_t)NDIR2 * sizeof(float));
  float* Z    = (float*)walloc((size_t)NTOK * NDIR2 * sizeof(float)); // 134 MB
  float* zr   = (float*)walloc((size_t)2 * NB * G4H * sizeof(float)); // 2 MB
  bf16*  hbuf = (bf16*) walloc((size_t)2 * NB * HDIM * sizeof(bf16));
  float* cbuf = (float*)walloc((size_t)2 * NB * HDIM * sizeof(float));

  // init states + output running max
  k_init<<<(2 * NB * HDIM + 255) / 256, 256, 0, stream>>>(hbuf, cbuf, out);

  // convert weights/bias
  int convN = NDIR2 * EPAD + NDIR2 * HDIM + NDIR2;
  k_convert<<<(convN + 255) / 256, 256, 0, stream>>>(wih_f, whh_f, b_f,
                                                     wih_b, whh_b, b_b,
                                                     Wih, Whh, bias);
  // embedding gather -> bf16 time-major
  k_embed<<<NTOK, 128, 0, stream>>>(tok, emb, X);

  // big parallel input-projection GEMM (both directions fused on N)
  k_inproj<<<4096, 256, 0, stream>>>(X, Wih, bias, Z);

  // serial scan: per-step recurrent GEMM + fused gates/max
  for (int t = 0; t < NT; ++t) {
    k_recur_gemm<<<64, 256, 0, stream>>>(hbuf, Whh, Z, zr, t);
    k_gates<<<(2 * NB * HDIM + 511) / 512, 512, 0, stream>>>(zr, lens, cbuf, hbuf, out, t);
  }
}
